// RPN_79336635891801
// MI455X (gfx1250) — matrix operand order repaired and checked by reading, not compiled, not run on hardware
//
#include <hip/hip_runtime.h>

// ---------------------------------------------------------------- constants
#define RH 200
#define RW 272
#define RHW (RH * RW)          // 54400 spatial positions
#define NANCH (RHW * 9)        // 489600 anchors/scores
#define PRE 6000
#define POST 1000
#define MASKW 188              // ceil(6000/32)

typedef __attribute__((ext_vector_type(16))) __bf16 bfx16;
typedef __attribute__((ext_vector_type(8)))  float  v8f;
typedef __attribute__((ext_vector_type(4)))  unsigned u32x4;
typedef __attribute__((ext_vector_type(8)))  unsigned u32x8;

union ABfrag { bfx16 v; uint4 q[2]; };

__device__ inline __bf16 f2bf(float f) {
  unsigned u = __float_as_uint(f);
  unsigned r = (u + 0x7FFFu + ((u >> 16) & 1u)) >> 16;   // round-to-nearest-even
  unsigned short s = (unsigned short)r;
  __bf16 out;
  __builtin_memcpy(&out, &s, 2);
  return out;
}

__device__ inline unsigned fkey(float f) {   // monotone float -> uint
  unsigned u = __float_as_uint(f);
  return u ^ ((u & 0x80000000u) ? 0xFFFFFFFFu : 0x80000000u);
}

// ---------------------------------------------------------------- TDM helper
// 2-D tensor_load_to_lds: tile of `tw` bf16 elements x `th_` rows, row stride
// `stride_elems`, packed contiguously into LDS at byte offset `lds_off`.
// D# group0: count=1 | lds_addr | global_addr[56:0] | type=2.
// D# group1: data_size=2B, huge tensor dims (tile known in-bounds), tile dims,
// dim0 stride. 2-group form (VADDR2/3 = NULL) => tensors up to 2D.
__device__ inline void tdm_load_2d_bf16(const void* gptr, unsigned lds_off,
                                        unsigned tw, unsigned th_,
                                        unsigned stride_elems) {
  unsigned long long ga = (unsigned long long)gptr;
  u32x4 g0;
  g0[0] = 1u;                                   // count=1, user mode
  g0[1] = lds_off;                              // lds_addr (bytes)
  g0[2] = (unsigned)ga;                         // global_addr[31:0]
  g0[3] = ((unsigned)(ga >> 32) & 0x01FFFFFFu)  // global_addr[56:32]
          | (2u << 30);                         // type = 2 ("image")
  const unsigned td0 = 0x40000000u, td1 = 0x40000000u;  // huge: no OOB clipping
  u32x8 g1;
  g1[0] = 1u << 16;                             // data_size = 1 -> 2 bytes
  g1[1] = (td0 & 0xFFFFu) << 16;                // tensor_dim0 lo16
  g1[2] = (td0 >> 16) | ((td1 & 0xFFFFu) << 16);
  g1[3] = (td1 >> 16) | (tw << 16);             // tile_dim0
  g1[4] = th_;                                  // tile_dim1 (tile_dim2 = 0)
  g1[5] = stride_elems;                         // tensor_dim0_stride lo32
  g1[6] = 0u;
  g1[7] = 0u;
  asm volatile("tensor_load_to_lds %0, %1" :: "s"(g0), "s"(g1) : "memory");
}

// --------------------- K0: tiled transpose feature [c][s] -> bf16 [s][c]
__global__ __launch_bounds__(256)
void k_transpose_feat(const float* __restrict__ feat,
                      __bf16* __restrict__ featT) {
  __shared__ __align__(16) __bf16 T[64 * 256];   // [si][ci], 32 KB
  const int tid = threadIdx.x;
  const int s0 = blockIdx.x * 64;
#pragma unroll 8
  for (int e = 0; e < 64; ++e) {
    int flat = e * 256 + tid;       // 0..16383
    int ci = flat >> 6;             // 0..255
    int si = flat & 63;             // coalesced along spatial
    T[si * 256 + ci] = f2bf(feat[(size_t)ci * RHW + s0 + si]);
  }
  __syncthreads();
#pragma unroll
  for (int e = 0; e < 8; ++e) {
    int flat = e * 256 + tid;       // 0..2047
    int si = flat >> 5;
    int g = flat & 31;              // 8-element group within channel dim
    *(uint4*)&featT[(size_t)(s0 + si) * 256 + g * 8] =
        *(const uint4*)&T[si * 256 + g * 8];
  }
}

// --------------------------------- K1: repack weights into WMMA-friendly bf16
// wpackT[tap][o][c]  (3x3 conv weights, N-major rows are TDM tile rows)
// wcrT [n][c]        (cls 0..8, reg 9..44, pad 45..47) for the 1x1 GEMM
__global__ void k_pack_w(const float* __restrict__ w1,
                         const float* __restrict__ wcls,
                         const float* __restrict__ wreg,
                         __bf16* __restrict__ wpackT,
                         __bf16* __restrict__ wcrT) {
  const int NW = 9 * 256 * 256;
  const int NT = NW + 48 * 256;
  for (int id = blockIdx.x * blockDim.x + threadIdx.x; id < NT;
       id += gridDim.x * blockDim.x) {
    if (id < NW) {
      int t = id >> 16;              // tap
      int rem = id & 0xFFFF;
      int o = rem >> 8, c = rem & 255;
      wpackT[id] = f2bf(w1[o * 2304 + c * 9 + t]);
    } else {
      int j = id - NW;
      int n = j >> 8, c = j & 255;
      float v = 0.f;
      if (n < 9)       v = wcls[n * 256 + c];
      else if (n < 45) v = wreg[(n - 9) * 256 + c];
      wcrT[j] = f2bf(v);
    }
  }
}

// ------------------------------- K2: 3x3 conv as implicit GEMM (bf16 WMMA)
// Block = 256 thr (8 waves). Tile: M=64 spatial x N=256 out-channels.
// Double-buffered LDS, one barrier per K-step. A staged as one b128/thread;
// B tile (16 KB, workgroup-uniform) fetched by the Tensor Data Mover one
// step ahead, drained with s_wait_tensorcnt before the barrier.
__global__ __launch_bounds__(256)
void k_conv3x3_wmma(const __bf16* __restrict__ featT,   // [s][c]
                    const __bf16* __restrict__ wpackT,  // [tap][o][c]
                    const float* __restrict__ b1,
                    __bf16* __restrict__ xbf) {         // [s][o]
  __shared__ __align__(16) __bf16 As[2][64 * 32];    // [m][k]
  __shared__ __align__(16) __bf16 Bs[2][256 * 32];   // [n][k]

  const int tid  = threadIdx.x;
  const int lane = tid & 31;
  const int wave = tid >> 5;
  const int wm   = wave & 3;
  const int wn   = wave >> 2;
  const int hl   = lane >> 4;     // K-half select per WMMA layout
  const int nl   = lane & 15;
  const int mbase = blockIdx.x * 64;
  // scalar (wave-uniform) wave-0 predicate: TDM ignores EXEC, so gate with a
  // uniform branch, not a lane mask.
  const bool w0 = (__builtin_amdgcn_readfirstlane((int)tid) >> 5) == 0;

  v8f acc[8];
#pragma unroll
  for (int t = 0; t < 8; ++t)
    acc[t] = (v8f){0.f, 0.f, 0.f, 0.f, 0.f, 0.f, 0.f, 0.f};

  // A-staging role: row am, 8 channels starting at ak (one b128 per step)
  const int am = tid >> 2;
  const int ak = (tid & 3) * 8;
  const int sa = mbase + am;
  const int ay = sa / RW, ax = sa % RW;

  auto loadA = [&](int t, uint4& r) {
    const int tap = t >> 3, cb = t & 7;
    const int dy = tap / 3 - 1, dx = tap % 3 - 1;
    const int yy = ay + dy, xx = ax + dx;
    r.x = 0u; r.y = 0u; r.z = 0u; r.w = 0u;
    if (yy >= 0 && yy < RH && xx >= 0 && xx < RW)
      r = *(const uint4*)&featT[(size_t)(yy * RW + xx) * 256 + cb * 32 + ak];
  };
  auto issueB = [&](int t, int buf) {
    const int tap = t >> 3, cb = t & 7;
    tdm_load_2d_bf16(&wpackT[((size_t)tap * 256) * 256 + cb * 32],
                     (unsigned)(size_t)&Bs[buf][0],
                     32u, 256u, 256u);
  };

  // prologue: stage step 0 into buffer 0
  {
    uint4 r;
    loadA(0, r);
    *(uint4*)&As[0][am * 32 + ak] = r;
  }
  if (w0) {
    issueB(0, 0);
    __builtin_amdgcn_s_wait_tensorcnt(0);
  }
  __syncthreads();

  for (int t = 0; t < 72; ++t) {           // 9 taps x 8 channel blocks
    const int cur = t & 1, nxt = cur ^ 1;
    const bool more = (t + 1) < 72;
    uint4 rA;
    if (more) {
      if (w0) issueB(t + 1, nxt);          // DMA next B tile under the WMMAs
      loadA(t + 1, rA);
    }

    ABfrag a;   // A 16x32 bf16: elems 0..7 -> k=hl*8+e ; 8..15 -> 16+hl*8+e
    a.q[0] = *(const uint4*)&As[cur][(wm * 16 + nl) * 32 + hl * 8];
    a.q[1] = *(const uint4*)&As[cur][(wm * 16 + nl) * 32 + 16 + hl * 8];
#pragma unroll
    for (int t16 = 0; t16 < 8; ++t16) {
      ABfrag b; // B 32x16 bf16: lane n = nl, elems 0..7 -> k=hl*8+e ; 8..15 -> 16+hl*8+e
      const int n = wn * 128 + t16 * 16 + nl;
      b.q[0] = *(const uint4*)&Bs[cur][n * 32 + hl * 8];
      b.q[1] = *(const uint4*)&Bs[cur][n * 32 + 16 + hl * 8];
      acc[t16] = __builtin_amdgcn_wmma_f32_16x16x32_bf16(
          false, a.v, false, b.v, (short)0, acc[t16], false, false);
    }

    if (more) {
      *(uint4*)&As[nxt][am * 32 + ak] = rA;
      if (w0) __builtin_amdgcn_s_wait_tensorcnt(0);
    }
    __syncthreads();
  }

  // epilogue: bias + ReLU, store x as [spatial][channel] bf16 for the 1x1 GEMM
#pragma unroll
  for (int t16 = 0; t16 < 8; ++t16) {
    const int o = wn * 128 + t16 * 16 + nl;
    const float bias = b1[o];
#pragma unroll
    for (int r = 0; r < 8; ++r) {
      const int m = wm * 16 + hl * 8 + r;
      float v = acc[t16][r] + bias;
      v = v > 0.f ? v : 0.f;
      xbf[(size_t)(mbase + m) * 256 + o] = f2bf(v);
    }
  }
}

// ---------------------- K3: fused 1x1 convs (cls+reg) as GEMM (bf16 WMMA)
// M=54400, K=256, N=48 (9 cls + 36 reg + 3 pad). Block = 128 M rows, 8 waves.
// Whole 24 KB B matrix preloaded once by the TDM.
__global__ __launch_bounds__(256)
void k_conv1x1_wmma(const __bf16* __restrict__ xbf,
                    const __bf16* __restrict__ wcrT,
                    const float* __restrict__ bcls,
                    const float* __restrict__ breg,
                    float* __restrict__ conf,
                    float* __restrict__ offs) {
  __shared__ __align__(16) __bf16 Bs[48 * 256];   // [n][k]
  const int tid  = threadIdx.x;
  const int lane = tid & 31;
  const int wave = tid >> 5;
  const int hl   = lane >> 4;
  const int nl   = lane & 15;
  const int mbase = blockIdx.x * 128 + wave * 16;

  if ((__builtin_amdgcn_readfirstlane((int)tid) >> 5) == 0) {
    tdm_load_2d_bf16(wcrT, (unsigned)(size_t)&Bs[0], 12288u, 1u, 12288u);
    __builtin_amdgcn_s_wait_tensorcnt(0);
  }
  __syncthreads();

  v8f acc[3];
#pragma unroll
  for (int t = 0; t < 3; ++t)
    acc[t] = (v8f){0.f, 0.f, 0.f, 0.f, 0.f, 0.f, 0.f, 0.f};

  const int m = mbase + nl;
  for (int cb = 0; cb < 8; ++cb) {
    ABfrag a;
    a.q[0] = *(const uint4*)&xbf[(size_t)m * 256 + cb * 32 + hl * 8];
    a.q[1] = *(const uint4*)&xbf[(size_t)m * 256 + cb * 32 + 16 + hl * 8];
#pragma unroll
    for (int nb = 0; nb < 3; ++nb) {
      ABfrag b;
      b.q[0] = *(const uint4*)&Bs[(nb * 16 + nl) * 256 + cb * 32 + hl * 8];
      b.q[1] = *(const uint4*)&Bs[(nb * 16 + nl) * 256 + cb * 32 + 16 + hl * 8];
      acc[nb] = __builtin_amdgcn_wmma_f32_16x16x32_bf16(
          false, a.v, false, b.v, (short)0, acc[nb], false, false);
    }
  }

#pragma unroll
  for (int nb = 0; nb < 3; ++nb) {
    const int n = nb * 16 + nl;
#pragma unroll
    for (int r = 0; r < 8; ++r) {
      const int s = mbase + hl * 8 + r;
      float v = acc[nb][r];
      if (n < 9)       conf[(size_t)s * 9 + n] = v + bcls[n];
      else if (n < 45) offs[(size_t)s * 36 + (n - 9)] = v + breg[n - 9];
    }
  }
}

// ---------------------------------------- K4: exact radix top-6000 selection
// meta: [0]=prefix/key [1]=unused [2]=remaining need [3]=count strictly greater
//       [4]=cnt_hi [5]=cnt_eq
__global__ void k_zero_sel(unsigned* hist, int* meta) {
  int i = blockIdx.x * blockDim.x + threadIdx.x;
  if (i < 2048) hist[i] = 0;
  if (i < 16)   meta[i] = (i == 2) ? PRE : 0;
}

__global__ void k_hist_r(const float* __restrict__ conf, unsigned* hist,
                         const int* meta, unsigned fixmask, int shift, int nb) {
  const unsigned prefix = (unsigned)meta[0];
  for (int i = blockIdx.x * blockDim.x + threadIdx.x; i < NANCH;
       i += gridDim.x * blockDim.x) {
    unsigned key = fkey(conf[i]);
    if ((key & fixmask) == (prefix & fixmask))
      atomicAdd(&hist[(key >> shift) & (unsigned)(nb - 1)], 1u);
  }
}

__global__ void k_scan_r(unsigned* hist, int* meta, int shift, int nb) {
  if (threadIdx.x == 0) {
    unsigned remaining = (unsigned)meta[2];
    unsigned cum = 0;
    int d = nb - 1;
    for (; d > 0; --d) {
      if (cum + hist[d] >= remaining) break;
      cum += hist[d];
    }
    meta[3] += (int)cum;
    meta[2]  = (int)(remaining - cum);
    meta[0] |= d << shift;
  }
  __syncthreads();
  for (int i = threadIdx.x; i < 2048; i += blockDim.x) hist[i] = 0;
}

__global__ void k_compact_sel(const float* __restrict__ conf, int* meta,
                              float* selsc, int* selix) {
  const unsigned kstar = (unsigned)meta[0];
  const int G = meta[3], E = meta[2];
  for (int i = blockIdx.x * blockDim.x + threadIdx.x; i < NANCH;
       i += gridDim.x * blockDim.x) {
    float sc = conf[i];
    unsigned key = fkey(sc);
    int slot = -1;
    if (key > kstar) slot = atomicAdd(&meta[4], 1);
    else if (key == kstar) {
      int q = atomicAdd(&meta[5], 1);
      if (q < E) slot = G + q;
    }
    if (slot >= 0 && slot < PRE) { selsc[slot] = sc; selix[slot] = i; }
  }
}

// ----------------------- K5: one-block bitonic sort of 6000 (pad to 8192)
__global__ __launch_bounds__(1024)
void k_sort(const float* __restrict__ selsc, const int* __restrict__ selix,
            float* ssc, int* six) {
  __shared__ float ks[8192];
  __shared__ int   vs[8192];
  const int tid = threadIdx.x;
  for (int i = tid; i < 8192; i += 1024) {
    if (i < PRE) { ks[i] = selsc[i]; vs[i] = selix[i]; }
    else         { ks[i] = -3.0e38f; vs[i] = 0x7FFFFFFF; }
  }
  __syncthreads();
  for (int k = 2; k <= 8192; k <<= 1) {
    for (int j = k >> 1; j > 0; j >>= 1) {
      for (int i = tid; i < 8192; i += 1024) {
        int l = i ^ j;
        if (l > i) {
          bool up = ((i & k) == 0);
          float ki = ks[i], kl = ks[l];
          int vi = vs[i], vl = vs[l];
          bool ib = (ki > kl) || (ki == kl && vi < vl);  // i "better"
          if (ib != up) { ks[i] = kl; ks[l] = ki; vs[i] = vl; vs[l] = vi; }
        }
      }
      __syncthreads();
    }
  }
  for (int i = tid; i < PRE; i += 1024) { ssc[i] = ks[i]; six[i] = vs[i]; }
}

// -------------------------- K6: anchor gen + box decode + clip + min-size
__global__ void k_decode(const int* __restrict__ six,
                         const float* __restrict__ offs,
                         const int* __restrict__ pih,
                         const int* __restrict__ piw,
                         float* boxes, int* valid) {
  const int i = blockIdx.x * blockDim.x + threadIdx.x;
  if (i >= PRE) return;
  int idx = six[i];
  float bx0 = 0.f, bx1 = 0.f, bx2 = 0.f, bx3 = 0.f;
  int ok = 0;
  if (idx >= 0 && idx < NANCH) {
    int a = idx % 9, cell = idx / 9;
    int x = cell % RW, y = cell / RW;
    int si = a / 3, ri = a % 3;
    const float SC[3] = {32.f, 64.f, 128.f};
    const float RT[3] = {0.5f, 1.f, 2.f};
    float r = RT[ri], s = SC[si];
    float wa = s * sqrtf(1.f / r), ha = s * sqrtf(r);
    float cxa = (x + 0.5f) * 4.f, cya = (y + 0.5f) * 4.f;
    const float* d = &offs[(size_t)cell * 36 + a * 4];
    float cx = d[0] * wa + cxa, cy = d[1] * ha + cya;
    float ww = __expf(d[2]) * wa, hh = __expf(d[3]) * ha;
    const float IW = (float)piw[0], IH = (float)pih[0];
    bx0 = fminf(fmaxf(cx - 0.5f * ww, 0.f), IW);
    bx1 = fminf(fmaxf(cy - 0.5f * hh, 0.f), IH);
    bx2 = fminf(fmaxf(cx + 0.5f * ww, 0.f), IW);
    bx3 = fminf(fmaxf(cy + 0.5f * hh, 0.f), IH);
    ok = ((bx2 - bx0) >= 1.f) && ((bx3 - bx1) >= 1.f);
  }
  boxes[i * 4 + 0] = bx0; boxes[i * 4 + 1] = bx1;
  boxes[i * 4 + 2] = bx2; boxes[i * 4 + 3] = bx3;
  valid[i] = ok;
}

// --------------------------------------------- K7: IoU suppression bitmask
__global__ void k_mask(const float* __restrict__ boxes, unsigned* mask) {
  const int total = PRE * MASKW;
  const float4* b4 = (const float4*)boxes;
  for (int id = blockIdx.x * blockDim.x + threadIdx.x; id < total;
       id += gridDim.x * blockDim.x) {
    const int i = id / MASKW, w = id % MASKW;
    const float4 a = b4[i];
    const float area_a = (a.z - a.x) * (a.w - a.y);
    unsigned bits = 0;
    for (int b = 0; b < 32; ++b) {
      const int j = w * 32 + b;
      if (j >= PRE) break;
      const float4 q = b4[j];
      const float area_b = (q.z - q.x) * (q.w - q.y);
      const float lx = fmaxf(a.x, q.x), ly = fmaxf(a.y, q.y);
      const float rx = fminf(a.z, q.z), ry = fminf(a.w, q.w);
      const float iw = fmaxf(rx - lx, 0.f), ih = fmaxf(ry - ly, 0.f);
      const float inter = iw * ih;
      const float iou = inter / (area_a + area_b - inter + 1e-9f);
      if (iou > 0.7f) bits |= 1u << b;
    }
    mask[(size_t)i * MASKW + w] = bits;
  }
}

// ----------------- K8: sequential greedy NMS + compaction to d_out (1000x4)
__global__ __launch_bounds__(256)
void k_nms_out(const unsigned* __restrict__ mask,
               const int* __restrict__ valid,
               const float* __restrict__ boxes, float* out) {
  __shared__ unsigned keepw[MASKW];
  __shared__ int flag;
  const int tid = threadIdx.x;
  for (int i = tid; i < MASKW; i += 256) keepw[i] = 0;
  for (int i = tid; i < POST * 4; i += 256) out[i] = 0.f;
  __syncthreads();
  for (int i = 0; i < PRE; ++i) {
    if (tid == 0) flag = 0;
    __syncthreads();
    const unsigned* row = &mask[(size_t)i * MASKW];
    int f = 0;
    for (int w = tid; w < MASKW; w += 256)
      if (row[w] & keepw[w]) f = 1;
    if (f) flag = 1;                 // benign race, all write 1
    __syncthreads();
    if (tid == 0 && valid[i] && !flag) keepw[i >> 5] |= 1u << (i & 31);
    __syncthreads();
  }
  if (tid == 0) {
    int r = 0;
    for (int i = 0; i < PRE && r < POST; ++i) {
      if (keepw[i >> 5] & (1u << (i & 31))) {
        out[r * 4 + 0] = boxes[i * 4 + 0];
        out[r * 4 + 1] = boxes[i * 4 + 1];
        out[r * 4 + 2] = boxes[i * 4 + 2];
        out[r * 4 + 3] = boxes[i * 4 + 3];
        ++r;
      }
    }
  }
}

// --------------------------------------------------------------- launcher
extern "C" void kernel_launch(void* const* d_in, const int* in_sizes, int n_in,
                              void* d_out, int out_size, void* d_ws,
                              size_t ws_size, hipStream_t stream) {
  const float* feature = (const float*)d_in[0];
  const float* w1      = (const float*)d_in[1];
  const float* b1      = (const float*)d_in[2];
  const float* w_cls   = (const float*)d_in[3];
  const float* b_cls   = (const float*)d_in[4];
  const float* w_reg   = (const float*)d_in[5];
  const float* b_reg   = (const float*)d_in[6];
  const int*   p_ih    = (const int*)d_in[7];
  const int*   p_iw    = (const int*)d_in[8];
  float* out = (float*)d_out;

  char* ws = (char*)d_ws;
  size_t cur = 0;
  auto alloc = [&](size_t bytes) -> char* {
    char* p = ws + cur;
    cur = (cur + bytes + 255) & ~(size_t)255;
    return p;
  };
  __bf16* featT  = (__bf16*)alloc((size_t)RHW * 256 * 2);       // 27.9 MB
  __bf16* wpackT = (__bf16*)alloc((size_t)9 * 256 * 256 * 2);   // 1.2 MB
  __bf16* wcrT   = (__bf16*)alloc((size_t)48 * 256 * 2);
  __bf16* xbf    = (__bf16*)alloc((size_t)RHW * 256 * 2);       // 27.9 MB
  float*  conf   = (float*)alloc((size_t)NANCH * 4);            // 2.0 MB
  float*  offs   = (float*)alloc((size_t)RHW * 36 * 4);         // 7.8 MB
  unsigned* hist = (unsigned*)alloc(2048 * 4);
  int*    meta   = (int*)alloc(16 * 4);
  float*  selsc  = (float*)alloc(PRE * 4);
  int*    selix  = (int*)alloc(PRE * 4);
  float*  ssc    = (float*)alloc(PRE * 4);
  int*    six    = (int*)alloc(PRE * 4);
  float*  boxes  = (float*)alloc(PRE * 4 * 4);
  int*    valid  = (int*)alloc(PRE * 4);
  unsigned* mask = (unsigned*)alloc((size_t)PRE * MASKW * 4);   // 4.5 MB

  // 1. transpose feature to [s][c] bf16 + weight repack
  k_transpose_feat<<<RHW / 64, 256, 0, stream>>>(feature, featT);
  k_pack_w<<<1024, 256, 0, stream>>>(w1, w_cls, w_reg, wpackT, wcrT);

  // 2. 3x3 conv + ReLU via bf16 WMMA + TDM-fed B tiles (~64 GFLOP)
  k_conv3x3_wmma<<<RHW / 64, 256, 0, stream>>>(featT, wpackT, b1, xbf);

  // 3. fused 1x1 cls/reg heads via bf16 WMMA (TDM-preloaded B)
  k_conv1x1_wmma<<<RHW / 128, 256, 0, stream>>>(xbf, wcrT, b_cls, b_reg,
                                                conf, offs);

  // 4. exact 32-bit radix select of top-6000 scores (deterministic)
  k_zero_sel<<<8, 256, 0, stream>>>(hist, meta);
  k_hist_r<<<2048, 256, 0, stream>>>(conf, hist, meta, 0x00000000u, 21, 2048);
  k_scan_r<<<1, 256, 0, stream>>>(hist, meta, 21, 2048);
  k_hist_r<<<2048, 256, 0, stream>>>(conf, hist, meta, 0xFFE00000u, 10, 2048);
  k_scan_r<<<1, 256, 0, stream>>>(hist, meta, 10, 2048);
  k_hist_r<<<2048, 256, 0, stream>>>(conf, hist, meta, 0xFFFFFC00u, 0, 1024);
  k_scan_r<<<1, 256, 0, stream>>>(hist, meta, 0, 1024);
  k_compact_sel<<<2048, 256, 0, stream>>>(conf, meta, selsc, selix);

  // 5. sort by (score desc, idx asc) — one-block bitonic in 64 KB LDS
  k_sort<<<1, 1024, 0, stream>>>(selsc, selix, ssc, six);

  // 6. decode + clip + min-size
  k_decode<<<(PRE + 255) / 256, 256, 0, stream>>>(six, offs, p_ih, p_iw,
                                                  boxes, valid);

  // 7. IoU bitmask, then sequential greedy NMS + compaction into d_out
  k_mask<<<(PRE * MASKW + 255) / 256, 256, 0, stream>>>(boxes, mask);
  k_nms_out<<<1, 256, 0, stream>>>(mask, valid, boxes, out);

  (void)in_sizes; (void)n_in; (void)out_size; (void)ws_size;
}
